// AlignmentModel_50637664420428
// MI455X (gfx1250) — matrix operand order repaired and checked by reading, not compiled, not run on hardware
//
#include <hip/hip_runtime.h>

#define HH 1024
#define SS 2048
#define BB 32
#define KK 2048   // 2H

typedef __attribute__((ext_vector_type(16))) __bf16 v16bf;
typedef __attribute__((ext_vector_type(8)))  float  v8f;

__device__ __forceinline__ unsigned short f2bf(float x) {
  unsigned u = __float_as_uint(x);
  u += 0x7FFFu + ((u >> 16) & 1u);   // round-to-nearest-even
  return (unsigned short)(u >> 16);
}

// Hardware tanh (CDNA5 TRANS op, confirmed lowering to v_tanh_f32).
__device__ __forceinline__ float fast_tanh(float x) {
#if __has_builtin(__builtin_amdgcn_tanhf)
  return __builtin_amdgcn_tanhf(x);
#else
  float y;
  asm volatile("v_tanh_f32 %0, %1\n\tv_nop" : "=v"(y) : "v"(x));
  return y;
#endif
}

__global__ void zero_f32_k(float* __restrict__ p, int n) {
  int i = blockIdx.x * blockDim.x + threadIdx.x;
  if (i < n) p[i] = 0.f;
}

// Pack U_w [n=0..1023][k=0..2047] fp32 -> bf16 in WMMA B-fragment layout:
// [n/16][k/32][lane=(k_in/16)*16 + n%16][e = k_in%16]
__global__ __launch_bounds__(256) void pack_B_k(const float* __restrict__ Uw,
                                                unsigned short* __restrict__ Bb) {
  int idx = blockIdx.x * blockDim.x + threadIdx.x;   // over H * (KK/16) = 128K
  int n = idx >> 7;
  int k0 = (idx & 127) << 4;        // 16-aligned k chunk
  int nblk = n >> 4, kblk = k0 >> 5;
  int hi = (k0 >> 4) & 1;
  int lane = (n & 15) | (hi << 4);
  union { unsigned short s[16]; uint4 q[2]; } u;
  const float4* src = (const float4*)(Uw + (size_t)n * KK + k0);
  float4 f0 = src[0], f1 = src[1], f2 = src[2], f3 = src[3];
  const float f[16] = {f0.x,f0.y,f0.z,f0.w, f1.x,f1.y,f1.z,f1.w,
                       f2.x,f2.y,f2.z,f2.w, f3.x,f3.y,f3.z,f3.w};
#pragma unroll
  for (int j = 0; j < 16; ++j) u.s[j] = f2bf(f[j]);
  uint4* dst = (uint4*)(Bb + ((((size_t)nblk * (KK / 32) + kblk) * 32 + lane) << 4));
  dst[0] = u.q[0];
  dst[1] = u.q[1];
}

// wq[b,n] = dh[b,:]·W_w[n,:] + W_b[n] + U_b[n]; one wave per output.
__global__ __launch_bounds__(256) void wq_k(const float* __restrict__ dh,
                                            const float* __restrict__ Ww,
                                            const float* __restrict__ Wb,
                                            const float* __restrict__ Ub,
                                            float* __restrict__ wq) {
  int wid = (blockIdx.x * 256 + threadIdx.x) >> 5;   // output index, 0..B*H-1
  int lane = threadIdx.x & 31;
  int b = wid >> 10, n = wid & 1023;
  const float* h = dh + b * HH;
  const float* w = Ww + (size_t)n * HH;
  float acc = 0.f;
#pragma unroll 8
  for (int i = lane; i < HH; i += 32) acc += h[i] * w[i];
  acc += __shfl_xor(acc, 1);
  acc += __shfl_xor(acc, 2);
  acc += __shfl_xor(acc, 4);
  acc += __shfl_xor(acc, 8);
  acc += __shfl_xor(acc, 16);
  if (lane == 0) wq[wid] = acc + Wb[n] + Ub[n];
}

// Fused GEMM + tanh + V-dot partial-score kernel.
// Block = 256 threads (8 waves, 2x4), tile M=128, N=256, K-step 64.
// Each wave owns a 64x64 sub-tile = 4x4 WMMA accumulators.
__global__ __launch_bounds__(256) void attn_gemm_k(
    const float* __restrict__ enc, const unsigned short* __restrict__ Bb,
    const float* __restrict__ wq, const float* __restrict__ Vw,
    float* __restrict__ scores) {
  __shared__ __align__(32) unsigned short sA[2 * 128 * 32];   // 16 KB, 2 K-subtiles

  const int b   = blockIdx.z;
  const int s0  = blockIdx.x * 128;
  const int n0  = blockIdx.y * 256;
  const int tid = threadIdx.x;
  const int lane = tid & 31;
  const int wave = tid >> 5;
  const int wm = (wave & 1) * 64;     // wave row offset  (0 / 64)
  const int wn = (wave >> 1) * 64;    // wave col offset  (0 / 64 / 128 / 192)

  v8f z = {};
  v8f acc[4][4];
#pragma unroll
  for (int i = 0; i < 4; ++i)
#pragma unroll
    for (int j = 0; j < 4; ++j) acc[i][j] = z;

  // cooperative A-load: thread -> (row lm, 16-wide k half lh) per 32-k subtile
  const int lm = tid >> 1;            // 0..127
  const int lh = (tid & 1) * 16;
  const float* encRow = enc + ((size_t)b * SS + (s0 + lm)) * (size_t)KK + lh;

  // B fragment base pointers (ushort units), coalesced 32B/lane
  const int nb0 = (n0 + wn) >> 4;
  const unsigned short* bp[4];
#pragma unroll
  for (int j = 0; j < 4; ++j)
    bp[j] = Bb + ((size_t)(nb0 + j) * (KK / 32) * 32 + lane) * 16;

  // A fragment LDS bases: four 16-row groups per wave
  const int g0 = wm >> 4;
  const unsigned short* ap[4];
#pragma unroll
  for (int i = 0; i < 4; ++i)
    ap[i] = sA + (g0 + i) * 512 + lane * 16;

  const int gq = lm >> 4;
  const int mq = lm & 15;

  for (int kt = 0; kt < KK / 64; ++kt) {
    __syncthreads();
#pragma unroll
    for (int c = 0; c < 2; ++c) {
      union { float4 q[4]; float f[16]; } u;
      const float4* src = (const float4*)(encRow + kt * 64 + c * 32);
      u.q[0] = src[0]; u.q[1] = src[1]; u.q[2] = src[2]; u.q[3] = src[3];
#pragma unroll
      for (int j = 0; j < 16; ++j) {
        int kl = lh + j;                                   // 0..31 within subtile
        int hi = kl >> 4;
        int e  = kl & 15;
        sA[c * 4096 + gq * 512 + ((mq | (hi << 4)) << 4) + e] = f2bf(u.f[j]);
      }
    }
    __syncthreads();
#pragma unroll
    for (int c = 0; c < 2; ++c) {
      const int kt2 = kt * 2 + c;
      v16bf a[4], bf[4];
#pragma unroll
      for (int i = 0; i < 4; ++i) a[i]  = *(const v16bf*)(ap[i] + c * 4096);
#pragma unroll
      for (int j = 0; j < 4; ++j) bf[j] = *(const v16bf*)(bp[j] + kt2 * 512);
#pragma unroll
      for (int i = 0; i < 4; ++i)
#pragma unroll
        for (int j = 0; j < 4; ++j)
          acc[i][j] = __builtin_amdgcn_wmma_f32_16x16x32_bf16(
              false, a[i], false, bf[j], (short)0, acc[i][j], false, false);
    }
  }

  // Epilogue: partial score p[m] = sum_n Vw[n]*tanh(acc + wq[b,n])
  const int nl = lane & 15;
  const int hi = lane >> 4;
  float wqv[4], vwv[4];
#pragma unroll
  for (int j = 0; j < 4; ++j) {
    int nc = n0 + wn + j * 16 + nl;
    wqv[j] = wq[b * HH + nc];
    vwv[j] = Vw[nc];
  }

#pragma unroll
  for (int im = 0; im < 4; ++im) {
#pragma unroll
    for (int r = 0; r < 8; ++r) {
      float p = 0.f;
#pragma unroll
      for (int j = 0; j < 4; ++j)
        p += vwv[j] * fast_tanh(acc[im][j][r] + wqv[j]);
      p += __shfl_xor(p, 1);
      p += __shfl_xor(p, 2);
      p += __shfl_xor(p, 4);
      p += __shfl_xor(p, 8);
      if (nl == 0) {
        int srow = s0 + wm + im * 16 + r + hi * 8;
        atomicAdd(&scores[b * SS + srow], p);
      }
    }
  }
}

// in-place softmax over S per batch row
__global__ __launch_bounds__(256) void softmax_k(float* __restrict__ sc) {
  __shared__ float red[256];
  const int b = blockIdx.x;
  const int tid = threadIdx.x;
  float* row = sc + b * SS;

  float m = -1e30f;
  for (int i = tid; i < SS; i += 256) m = fmaxf(m, row[i]);
  red[tid] = m; __syncthreads();
  for (int off = 128; off > 0; off >>= 1) {
    if (tid < off) red[tid] = fmaxf(red[tid], red[tid + off]);
    __syncthreads();
  }
  m = red[0]; __syncthreads();

  float sum = 0.f;
  for (int i = tid; i < SS; i += 256) { float e = expf(row[i] - m); row[i] = e; sum += e; }
  red[tid] = sum; __syncthreads();
  for (int off = 128; off > 0; off >>= 1) {
    if (tid < off) red[tid] += red[tid + off];
    __syncthreads();
  }
  float inv = 1.f / red[0];
  for (int i = tid; i < SS; i += 256) row[i] *= inv;
}

// context[b,d] = sum_s weights[b,s] * enc[b,s,d]; float4 along d, split over S.
__global__ __launch_bounds__(256) void context_k(const float* __restrict__ enc,
                                                 const float* __restrict__ w,
                                                 float* __restrict__ out) {
  const int b = blockIdx.z;
  const int d4 = blockIdx.x * 256 + threadIdx.x;   // 0..511 float4 columns
  const int sb = blockIdx.y * 256;
  const float* wp = w + b * SS + sb;
  const float4* ep = (const float4*)(enc + ((size_t)b * SS + sb) * (size_t)KK) + d4;
  float4 acc = {0.f, 0.f, 0.f, 0.f};
#pragma unroll 4
  for (int s = 0; s < 256; ++s) {
    float ws = wp[s];
    float4 e = ep[(size_t)s * (KK / 4)];
    acc.x += ws * e.x; acc.y += ws * e.y; acc.z += ws * e.z; acc.w += ws * e.w;
  }
  float* o = out + b * KK + d4 * 4;
  atomicAdd(o + 0, acc.x);
  atomicAdd(o + 1, acc.y);
  atomicAdd(o + 2, acc.z);
  atomicAdd(o + 3, acc.w);
}

extern "C" void kernel_launch(void* const* d_in, const int* in_sizes, int n_in,
                              void* d_out, int out_size, void* d_ws, size_t ws_size,
                              hipStream_t stream) {
  const float* enc = (const float*)d_in[0];   // [B,S,2H]
  const float* dh  = (const float*)d_in[1];   // [1,B,H]
  const float* Ww  = (const float*)d_in[2];   // [H,H]
  const float* Wb  = (const float*)d_in[3];   // [H]
  const float* Uw  = (const float*)d_in[4];   // [H,2H]
  const float* Ub  = (const float*)d_in[5];   // [H]
  const float* Vw  = (const float*)d_in[6];   // [1,H]
  // d_in[7] (V_b) cancels under softmax -> unused
  float* out = (float*)d_out;                 // [B,1,2H] = 65536 floats

  // workspace layout
  unsigned short* Bb = (unsigned short*)d_ws;                       // 4 MB bf16 packed U_w
  float* wq = (float*)((char*)d_ws + (size_t)HH * KK * 2);          // 128 KB
  float* sc = wq + BB * HH;                                         // 256 KB scores/weights

  zero_f32_k<<<(BB * SS + 255) / 256, 256, 0, stream>>>(sc, BB * SS);
  zero_f32_k<<<(BB * KK + 255) / 256, 256, 0, stream>>>(out, BB * KK);
  pack_B_k<<<(HH * (KK / 16)) / 256, 256, 0, stream>>>(Uw, Bb);
  wq_k<<<(BB * HH * 32) / 256, 256, 0, stream>>>(dh, Ww, Wb, Ub, wq);

  dim3 gg(SS / 128, HH / 256, BB);   // 16 x 4 x 32 = 2048 blocks
  attn_gemm_k<<<gg, 256, 0, stream>>>(enc, Bb, wq, Vw, sc);

  softmax_k<<<BB, 256, 0, stream>>>(sc);

  dim3 gc(KK / 1024, SS / 256, BB);  // 2 x 8 x 32 blocks
  context_k<<<gc, 256, 0, stream>>>(enc, sc, out);
}
